// LearnedClassVectors_10385230922118
// MI455X (gfx1250) — matrix operand order repaired and checked by reading, not compiled, not run on hardware
//
#include <hip/hip_runtime.h>
#include <hip/hip_bf16.h>

typedef __attribute__((ext_vector_type(16))) _Float16 v16h;
typedef __attribute__((ext_vector_type(8)))  _Float16 v8h;
typedef __attribute__((ext_vector_type(8)))  float    v8f;

// Problem constants (from reference)
#define NPAT_TOT   65536      // 2 * 32*32*32 patches
#define MTILE      64         // patches per workgroup
#define KDIM       512        // V*vd = 64*8
#define NDIM       768        // out_dim
#define VROW       520        // padded LDS row stride in f16 (512 + 8)
#define NVEC       4608       // 9*64*8 floats

__global__ __launch_bounds__(256) void lcv_fused_kernel(
    const float* __restrict__ x,        // [2,1,128,128,128]
    const float* __restrict__ vectors,  // [9,64,8]
    const float* __restrict__ fc_w,     // [512,768]
    const float* __restrict__ fc_b,     // [768]
    float* __restrict__ out)            // [2,768,32,32,32]
{
    __shared__ _Float16 vvS[MTILE * VROW];  // 64 x 520 f16 = 66,560 B
    __shared__ float    vecs[NVEC];         // 18,432 B

    const int tid  = threadIdx.x;
    const int lane = tid & 31;
    const int wave = tid >> 5;
    const int l15  = lane & 15;
    const int half = lane >> 4;           // 0 or 1

    const size_t m0      = (size_t)blockIdx.x * MTILE;  // first patch of tile
    const int    bidx    = (int)(m0 >> 15);             // batch (32768 patches/batch)
    const size_t sp_base = m0 & 32767;                  // spatial flat index base

    // ---- stage interpolation vectors into LDS ----
    for (int i = tid; i < NVEC; i += 256) vecs[i] = vectors[i];
    __syncthreads();

    // ---- Phase A: embedding -> f16 LDS tile vv[p][k] ----
    // 64 patches * 64 voxels = 4096 tasks; p = t&63 so a wave writes
    // 32 distinct rows -> conflict-free b128 LDS stores.
    for (int it = 0; it < 16; ++it) {
        int t = tid + it * 256;
        int p = t & 63;          // patch within tile
        int v = t >> 6;          // voxel within patch (pd,ph,pw order)

        size_t m  = m0 + p;
        int    sp = (int)(m & 32767);
        int    bb = (int)(m >> 15);
        int pd = sp >> 10, ph = (sp >> 5) & 31, pw = sp & 31;
        int dz = v >> 4,  dy = (v >> 2) & 3,   dx = v & 3;
        size_t xi = (((size_t)bb * 128 + (size_t)(pd * 4 + dz)) * 128
                     + (size_t)(ph * 4 + dy)) * 128 + (size_t)(pw * 4 + dx);
        float val = x[xi];

        // searchsorted(HU, val, 'right') - 1, clipped to [0,7]
        int idx = (val >= -75.f) + (val >= 0.f)  + (val >= 15.f) + (val >= 25.f)
                + (val >= 40.f)  + (val >= 50.f) + (val >= 200.f);
        float lo = -1000.f, winv = 1.f / 925.f;
        if (val >= -75.f) { lo = -75.f; winv = 1.f / 75.f;  }
        if (val >= 0.f)   { lo = 0.f;   winv = 1.f / 15.f;  }
        if (val >= 15.f)  { lo = 15.f;  winv = 1.f / 10.f;  }
        if (val >= 25.f)  { lo = 25.f;  winv = 1.f / 15.f;  }
        if (val >= 40.f)  { lo = 40.f;  winv = 1.f / 10.f;  }
        if (val >= 50.f)  { lo = 50.f;  winv = 1.f / 150.f; }
        if (val >= 200.f) { lo = 200.f; winv = 1.f / 800.f; }
        float w = fminf(fmaxf((val - lo) * winv, 0.f), 1.f);

        const float* va = &vecs[(idx * 64 + v) * 8];   // vectors[idx , v, :]
        const float* vb = va + 512;                    // vectors[idx+1, v, :]
        v8h r;
        #pragma unroll
        for (int j = 0; j < 8; ++j)
            r[j] = (_Float16)((1.f - w) * va[j] + w * vb[j]);
        *(v8h*)&vvS[p * VROW + v * 8] = r;             // 16B aligned
    }
    __syncthreads();

    // ---- Phase B: D[n_ch][patch] = fc_w^T . vv^T via WMMA f32<-f16 ----
    // 48 N-tiles split over 8 waves; 4 M-subtiles of 16 patches each.
    for (int nt = wave; nt < 48; nt += 8) {
        const int n0 = nt * 16;
        v8f acc[4] = {};

        for (int kc = 0; kc < KDIM; kc += 32) {
            // A fragment (16x32 f16): A[m][kk] = fc_w[kc+kk][n0+m]
            // element e of lane -> kk = 16*half + e
            v16h a;
            #pragma unroll
            for (int e = 0; e < 16; ++e) {
                int kk = (half << 4) + e;
                a[e] = (_Float16)fc_w[(size_t)(kc + kk) * NDIM + n0 + l15];
            }
            #pragma unroll
            for (int ms = 0; ms < 4; ++ms) {
                // B fragment (32x16 f16): B[kk][n] = vv[ms*16+n][kc+kk]
                // element e of lane -> kk = 16*half + e  (contiguous run)
                const _Float16* bp =
                    &vvS[(ms * 16 + l15) * VROW + kc + (half << 4)];
                v8h blo = *(const v8h*)bp;
                v8h bhi = *(const v8h*)(bp + 8);
                v16h b = __builtin_shufflevector(blo, bhi,
                    0,1,2,3,4,5,6,7,8,9,10,11,12,13,14,15);
                acc[ms] = __builtin_amdgcn_wmma_f32_16x16x32_f16(
                    false, a, false, b, (short)0, acc[ms], false, false);
            }
        }

        // epilogue: D row r -> channel n0 + r + 8*half ; lane -> patch
        float bias[8];
        #pragma unroll
        for (int r = 0; r < 8; ++r)
            bias[r] = fc_b[n0 + r + (half << 3)];

        #pragma unroll
        for (int ms = 0; ms < 4; ++ms) {
            size_t sp = sp_base + (size_t)(ms * 16 + l15);
            #pragma unroll
            for (int r = 0; r < 8; ++r) {
                int ch = n0 + r + (half << 3);
                out[((size_t)bidx * NDIM + ch) * 32768 + sp] = acc[ms][r] + bias[r];
            }
        }
    }
}

extern "C" void kernel_launch(void* const* d_in, const int* in_sizes, int n_in,
                              void* d_out, int out_size, void* d_ws, size_t ws_size,
                              hipStream_t stream) {
    const float* x       = (const float*)d_in[0];   // [2,1,128,128,128]
    const float* vectors = (const float*)d_in[1];   // [9,64,8]
    const float* fc_w    = (const float*)d_in[2];   // [512,768]
    const float* fc_b    = (const float*)d_in[3];   // [768]
    float* out           = (float*)d_out;           // [2,768,32,32,32]

    (void)in_sizes; (void)n_in; (void)out_size; (void)d_ws; (void)ws_size;

    dim3 grid(NPAT_TOT / MTILE);   // 1024 workgroups, one 64-patch tile each
    dim3 block(256);               // 8 wave32 per WG
    lcv_fused_kernel<<<grid, block, 0, stream>>>(x, vectors, fc_w, fc_b, out);
}
